// TransformerAttention_24859270709884
// MI455X (gfx1250) — matrix operand order repaired and checked by reading, not compiled, not run on hardware
//
#include <hip/hip_runtime.h>

// ---------------------------------------------------------------------------
// TransformerAttention for MI455X (gfx1250, CDNA5, wave32)
// bf16 WMMA (v_wmma_f32_16x16x32_bf16) everywhere, f32 accumulation,
// flash-attention streaming softmax (never materializes the 1GB score tensor).
// ---------------------------------------------------------------------------

#define BATCH  4
#define SEQ    2048
#define DIM    1024
#define HEADS  16
#define DH     64
#define PREFIX 1024

typedef __attribute__((ext_vector_type(16))) __bf16 v16bf;
typedef __attribute__((ext_vector_type(8)))  __bf16 v8bf;
typedef __attribute__((ext_vector_type(8)))  float  v8f;

__device__ __forceinline__ v8f wmma_bf16(v16bf a, v16bf b, v8f c) {
  // (neg_a, A, neg_b, B, c_mod, C, reuse_a, reuse_b)
  return __builtin_amdgcn_wmma_f32_16x16x32_bf16(false, a, false, b, (short)0, c,
                                                 false, false);
}

__device__ __forceinline__ __bf16 f2bf(float f) {
  unsigned u = __builtin_bit_cast(unsigned, f);
  unsigned r = (u + 0x7FFFu + ((u >> 16) & 1u)) >> 16;  // round-to-nearest-even
  return __builtin_bit_cast(__bf16, (unsigned short)r);
}

// ---------------------------------------------------------------------------
// f32 -> bf16 conversion (vectorized by 4; all sizes are multiples of 4)
// ---------------------------------------------------------------------------
__global__ void ta_cvt_bf16(const float* __restrict__ in, __bf16* __restrict__ out,
                            int n) {
  int i = (blockIdx.x * blockDim.x + threadIdx.x) * 4;
  if (i + 3 < n) {
    float4 v = *(const float4*)(in + i);
    out[i + 0] = f2bf(v.x);
    out[i + 1] = f2bf(v.y);
    out[i + 2] = f2bf(v.z);
    out[i + 3] = f2bf(v.w);
  }
}

// ---------------------------------------------------------------------------
// Tiled bf16 GEMM: C[M,N] = A[M,K] @ B[K,N], A/B row-major bf16.
//   MODE 0: write bf16 row-major
//   MODE 1: write bf16 per-head-transposed  Kt[b][h][dh][SEQ]   (K projection)
//   MODE 2: write f32 row-major + bias                          (output proj)
// Workgroup: 256 threads (8 waves). Tile 128x128, K-step 32.
// Each wave: 16 rows x 128 cols = 8 accumulators of 16x16.
// ---------------------------------------------------------------------------
#define BM 128
#define BN 128
#define BK 32
#define AS_STRIDE 40    // halves; 80B rows: 16B aligned, conflict-free frag reads
#define BS_STRIDE 136   // halves; 272B rows: 16B aligned, staggered banks

template <int MODE>
__global__ __launch_bounds__(256) void ta_gemm_wmma(
    const __bf16* __restrict__ A, const __bf16* __restrict__ Bw,
    void* __restrict__ Cout, const float* __restrict__ bias, int M, int N, int K) {
  __shared__ __bf16 As[BM * AS_STRIDE];
  __shared__ __bf16 Bs[BK * BS_STRIDE];

  const int tid  = threadIdx.x;
  const int lane = tid & 31;
  const int wave = tid >> 5;

  const int m0  = blockIdx.y * BM;
  const int n0  = blockIdx.x * BN;
  const int wr0 = wave * 16;  // this wave's 16-row strip

  v8f zero = {};
  v8f acc[8];
#pragma unroll
  for (int t = 0; t < 8; ++t) acc[t] = zero;

  // cooperative global->LDS assignments
  const int arow = tid >> 1, acol = (tid & 1) * 16;  // A tile 128x32
  const int brow = tid >> 3, bcol = (tid & 7) * 16;  // B tile 32x128

  const int ar = wr0 + (lane & 15);     // A-fragment row
  const int kb = (lane >> 4) * 8;       // A-fragment K sub-offset

  for (int k0 = 0; k0 < K; k0 += BK) {
    __syncthreads();
    *(v16bf*)(&As[arow * AS_STRIDE + acol]) =
        *(const v16bf*)(A + (size_t)(m0 + arow) * K + k0 + acol);
    *(v16bf*)(&Bs[brow * BS_STRIDE + bcol]) =
        *(const v16bf*)(Bw + (size_t)(k0 + brow) * N + n0 + bcol);
    if (k0 + BK < K) {  // global_prefetch_b8 for the next K-slab
      __builtin_prefetch(A + (size_t)(m0 + arow) * K + k0 + BK + acol, 0, 1);
      __builtin_prefetch(Bw + (size_t)(k0 + BK + brow) * N + n0 + bcol, 0, 1);
    }
    __syncthreads();

    // A fragment (16x32), per ISA 16-bit A layout
    v8bf lo = *(const v8bf*)(&As[ar * AS_STRIDE + kb]);
    v8bf hi = *(const v8bf*)(&As[ar * AS_STRIDE + 16 + kb]);
    v16bf afrag;
#pragma unroll
    for (int e = 0; e < 8; ++e) { afrag[e] = lo[e]; afrag[8 + e] = hi[e]; }

#pragma unroll
    for (int t = 0; t < 8; ++t) {
      // B fragment (32x16): lane&15 = column, K rows in table order
      v16bf bfrag;
#pragma unroll
      for (int e = 0; e < 8; ++e) { bfrag[e] = Bs[(kb + e) * BS_STRIDE + t * 16 + (lane & 15)]; bfrag[8 + e] = Bs[(16 + kb + e) * BS_STRIDE + t * 16 + (lane & 15)]; }
      acc[t] = wmma_bf16(afrag, bfrag, acc[t]);
    }
  }

  // epilogue: C/D layout -> memory. VGPR r holds M = r (+8 for lanes 16-31),
  // column = lane & 15.
  const int rowadd = (lane >> 4) << 3;
  const int cbase  = lane & 15;
#pragma unroll
  for (int t = 0; t < 8; ++t) {
#pragma unroll
    for (int r = 0; r < 8; ++r) {
      const int gm = m0 + wr0 + rowadd + r;       // = b*SEQ + token
      const int gn = n0 + t * 16 + cbase;         // output feature
      const float vv = acc[t][r];
      if constexpr (MODE == 0) {
        ((__bf16*)Cout)[(size_t)gm * N + gn] = f2bf(vv);
      } else if constexpr (MODE == 1) {
        // Kt[((b*HEADS + h)*DH + c)][token] ; (h,c) = (gn>>6, gn&63)
        ((__bf16*)Cout)[((size_t)((gm >> 11) * DIM + gn)) * SEQ + (gm & (SEQ - 1))] =
            f2bf(vv);
      } else {
        ((float*)Cout)[(size_t)gm * N + gn] = vv + bias[gn];
      }
    }
  }
}

// ---------------------------------------------------------------------------
// Flash attention. Grid: (SEQ/64, BATCH*HEADS), 128 threads = 4 waves.
// Each wave owns 16 query rows, streams keys in blocks of 32.
// Visibility: j < max(PREFIX, i+1) -> contiguous key range per query tile.
// ---------------------------------------------------------------------------
__global__ __launch_bounds__(128) void ta_flash_attn(
    const __bf16* __restrict__ Q, const __bf16* __restrict__ Kt,
    const __bf16* __restrict__ V, __bf16* __restrict__ O) {
  __shared__ __bf16 Ps[4 * 16 * 32];  // per-wave 16x32 P staging (D-layout -> A-layout)

  const int tid  = threadIdx.x;
  const int lane = tid & 31;
  const int wave = tid >> 5;
  const int bh   = blockIdx.y;        // b*HEADS + h
  const int b    = bh >> 4;
  const int h    = bh & 15;
  const int i0   = blockIdx.x * 64 + wave * 16;

  const __bf16* Qb  = Q + ((size_t)b * SEQ) * DIM + h * DH;
  const __bf16* Ktb = Kt + (size_t)bh * DH * SEQ;      // [DH][SEQ] row-major
  const __bf16* Vb  = V + ((size_t)b * SEQ) * DIM + h * DH;
  __bf16*       Ob  = O + ((size_t)b * SEQ) * DIM + h * DH;

  const int r15    = lane & 15;
  const int kb     = (lane >> 4) * 8;
  const int rowadd = (lane >> 4) << 3;

  // Load the wave's Q fragments once: 16x64 = 2 fragments of 16x32
  v16bf qa[2];
#pragma unroll
  for (int c = 0; c < 2; ++c) {
    const __bf16* qr = Qb + (size_t)(i0 + r15) * DIM + c * 32;
    v8bf lo = *(const v8bf*)(qr + kb);
    v8bf hi = *(const v8bf*)(qr + 16 + kb);
#pragma unroll
    for (int e = 0; e < 8; ++e) { qa[c][e] = lo[e]; qa[c][8 + e] = hi[e]; }
  }

  v8f zero = {};
  v8f o[4];
#pragma unroll
  for (int n = 0; n < 4; ++n) o[n] = zero;
  float rm[8], rl[8];
#pragma unroll
  for (int r = 0; r < 8; ++r) { rm[r] = -1.0e30f; rl[r] = 0.0f; }

  const float scale = 0.125f;  // 1/sqrt(64)
  const int   jEnd  = (i0 + 16 > PREFIX) ? (i0 + 16) : PREFIX;
  __bf16*     Pw    = &Ps[wave * 16 * 32];

  for (int j0 = 0; j0 < jEnd; j0 += 32) {
    // ---- S = Q (16x64) @ K^T (64x32): two 16x16 tiles ----
    v8f st[2];
    st[0] = zero;
    st[1] = zero;
#pragma unroll
    for (int t = 0; t < 2; ++t) {
#pragma unroll
      for (int c = 0; c < 2; ++c) {
        v16bf kf;
#pragma unroll
        for (int e = 0; e < 8; ++e) { kf[e] = Ktb[(size_t)(c * 32 + kb + e) * SEQ + j0 + t * 16 + r15]; kf[8 + e] = Ktb[(size_t)(c * 32 + 16 + kb + e) * SEQ + j0 + t * 16 + r15]; }
        st[t] = wmma_bf16(qa[c], kf, st[t]);
      }
    }

    // ---- scale + mask (visible iff j < PREFIX or j <= i) ----
    float p[2][8];
#pragma unroll
    for (int t = 0; t < 2; ++t) {
      const int j = j0 + t * 16 + r15;
#pragma unroll
      for (int r = 0; r < 8; ++r) {
        const int i   = i0 + rowadd + r;
        const bool ok = (j < PREFIX) || (j <= i);
        p[t][r] = ok ? st[t][r] * scale : -1.0e30f;
      }
    }

    // ---- online softmax (rows live in 16-lane halves -> xor reductions) ----
#pragma unroll
    for (int r = 0; r < 8; ++r) {
      float tm = fmaxf(p[0][r], p[1][r]);
      tm = fmaxf(tm, __shfl_xor(tm, 1));
      tm = fmaxf(tm, __shfl_xor(tm, 2));
      tm = fmaxf(tm, __shfl_xor(tm, 4));
      tm = fmaxf(tm, __shfl_xor(tm, 8));
      const float nm   = fmaxf(rm[r], tm);
      const float corr = __expf(rm[r] - nm);
      rm[r] = nm;
      const float p0 = __expf(p[0][r] - nm);
      const float p1 = __expf(p[1][r] - nm);
      p[0][r] = p0;
      p[1][r] = p1;
      float rs = p0 + p1;
      rs += __shfl_xor(rs, 1);
      rs += __shfl_xor(rs, 2);
      rs += __shfl_xor(rs, 4);
      rs += __shfl_xor(rs, 8);
      rl[r] = rl[r] * corr + rs;
#pragma unroll
      for (int n = 0; n < 4; ++n) o[n][r] *= corr;
    }

    // ---- stage P (16x32 bf16) to LDS, re-read in A-fragment layout ----
#pragma unroll
    for (int t = 0; t < 2; ++t)
#pragma unroll
      for (int r = 0; r < 8; ++r)
        Pw[(rowadd + r) * 32 + t * 16 + r15] = f2bf(p[t][r]);

    v8bf plo = *(const v8bf*)(&Pw[r15 * 32 + kb]);
    v8bf phi = *(const v8bf*)(&Pw[r15 * 32 + 16 + kb]);
    v16bf pf;
#pragma unroll
    for (int e = 0; e < 8; ++e) { pf[e] = plo[e]; pf[8 + e] = phi[e]; }

    // ---- O += P (16x32) @ V (32x64) ----
#pragma unroll
    for (int n = 0; n < 4; ++n) {
      v16bf vf;
#pragma unroll
      for (int e = 0; e < 8; ++e) { vf[e] = Vb[(size_t)(j0 + kb + e) * DIM + n * 16 + r15]; vf[8 + e] = Vb[(size_t)(j0 + 16 + kb + e) * DIM + n * 16 + r15]; }
      o[n] = wmma_bf16(pf, vf, o[n]);
    }
  }

  // ---- normalize and write bf16 output (merged-head layout) ----
#pragma unroll
  for (int r = 0; r < 8; ++r) {
    const float inv = 1.0f / rl[r];
#pragma unroll
    for (int n = 0; n < 4; ++n) {
      Ob[(size_t)(i0 + rowadd + r) * DIM + n * 16 + r15] = f2bf(o[n][r] * inv);
    }
  }
}

// ---------------------------------------------------------------------------
// Launch: convert -> 3 projection GEMMs (K pre-transposed) -> flash attention
//         -> output GEMM (+bias, f32 to d_out).  Workspace ~72 MB.
// ---------------------------------------------------------------------------
extern "C" void kernel_launch(void* const* d_in, const int* in_sizes, int n_in,
                              void* d_out, int out_size, void* d_ws, size_t ws_size,
                              hipStream_t stream) {
  (void)in_sizes; (void)n_in; (void)out_size; (void)ws_size;

  const float* x  = (const float*)d_in[0];
  const float* Wq = (const float*)d_in[1];
  const float* Wk = (const float*)d_in[2];
  const float* Wv = (const float*)d_in[3];
  const float* Wo = (const float*)d_in[4];
  const float* bo = (const float*)d_in[5];

  char* ws = (char*)d_ws;
  const size_t MB = 1ull << 20;
  __bf16* xh  = (__bf16*)(ws + 0 * MB);   // 16 MB (reused as AO after projections)
  __bf16* Wqh = (__bf16*)(ws + 16 * MB);  // 2 MB
  __bf16* Wkh = (__bf16*)(ws + 18 * MB);  // 2 MB
  __bf16* Wvh = (__bf16*)(ws + 20 * MB);  // 2 MB
  __bf16* Woh = (__bf16*)(ws + 22 * MB);  // 2 MB
  __bf16* Qh  = (__bf16*)(ws + 24 * MB);  // 16 MB
  __bf16* Kt  = (__bf16*)(ws + 40 * MB);  // 16 MB  [B][H][DH][SEQ]
  __bf16* Vh  = (__bf16*)(ws + 56 * MB);  // 16 MB
  __bf16* AO  = xh;                       // attention output, reuses xh region

  const int nx = BATCH * SEQ * DIM;  // 8,388,608
  const int nw = DIM * DIM;          // 1,048,576
  ta_cvt_bf16<<<nx / 4 / 256, 256, 0, stream>>>(x, xh, nx);
  ta_cvt_bf16<<<nw / 4 / 256, 256, 0, stream>>>(Wq, Wqh, nw);
  ta_cvt_bf16<<<nw / 4 / 256, 256, 0, stream>>>(Wk, Wkh, nw);
  ta_cvt_bf16<<<nw / 4 / 256, 256, 0, stream>>>(Wv, Wvh, nw);
  ta_cvt_bf16<<<nw / 4 / 256, 256, 0, stream>>>(Wo, Woh, nw);

  const int M = BATCH * SEQ;  // 8192
  dim3 gg(DIM / BN, M / BM);  // (8, 64)
  ta_gemm_wmma<0><<<gg, 256, 0, stream>>>(xh, Wqh, (void*)Qh, nullptr, M, DIM, DIM);
  ta_gemm_wmma<1><<<gg, 256, 0, stream>>>(xh, Wkh, (void*)Kt, nullptr, M, DIM, DIM);
  ta_gemm_wmma<0><<<gg, 256, 0, stream>>>(xh, Wvh, (void*)Vh, nullptr, M, DIM, DIM);

  dim3 ga(SEQ / 64, BATCH * HEADS);  // (32, 64)
  ta_flash_attn<<<ga, 128, 0, stream>>>(Qh, Kt, Vh, AO);

  ta_gemm_wmma<2><<<gg, 256, 0, stream>>>(AO, Woh, d_out, bo, M, DIM, DIM);
}
